// PersistentHomologyLoss_81003083202855
// MI455X (gfx1250) — matrix operand order repaired and checked by reading, not compiled, not run on hardware
//
#include <hip/hip_runtime.h>
#include <cstdint>
#include <cstddef>

// ---------------- types ----------------
typedef __attribute__((ext_vector_type(16))) _Float16 v16h;
typedef __attribute__((ext_vector_type(8)))  _Float16 v8h;
typedef __attribute__((ext_vector_type(4)))  _Float16 v4h;
typedef __attribute__((ext_vector_type(8)))  float    v8f;
typedef __attribute__((ext_vector_type(4)))  unsigned v4u;
typedef __attribute__((ext_vector_type(8)))  int      v8i;
typedef __attribute__((ext_vector_type(4)))  int      v4i;

#define N_PTS   1024
#define PDIM    128
#define NBATCH  128

#ifndef USE_TDM
#define USE_TDM 1
#endif

// LDS carve (bytes). Planes: 128 rows x 136 halves (pad -> bank spread).
#define LDSTRIDE 136
#define AHI_OFF  0
#define ALO_OFF  34816
#define BHI_OFF  69632
#define BLO_OFF  104448
#define STG_OFF  69632      // fp32 stage (64KB) overlaps B planes; A is converted first
#define NA_OFF   139264
#define NB_OFF   139776
#define LDS_BYTES 140288

__device__ inline v16h cat8(v8h a, v8h b) {
  return __builtin_shufflevector(a, b, 0,1,2,3,4,5,6,7,8,9,10,11,12,13,14,15);
}
__device__ inline v8f wmma_f16(v16h a, v16h b, v8f c) {
  return __builtin_amdgcn_wmma_f32_16x16x32_f16(false, a, false, b, (short)0, c, false, false);
}

// convert one 128x128 fp32 tile (row stride 128) into hi/lo f16 LDS planes
__device__ inline void planes_from_f32(const float* __restrict__ src, size_t src_stride,
                                       _Float16* hi, _Float16* lo, int tid) {
#pragma unroll
  for (int t = 0; t < 16; ++t) {
    int u = tid + (t << 8);                   // 0..4095 float4 slots
    int row = u >> 5, col = (u & 31) << 2;
    const float4 v = *(const float4*)(src + (size_t)row * src_stride + col);
    v4h h, l;
    h[0] = (_Float16)v.x; l[0] = (_Float16)(v.x - (float)h[0]);
    h[1] = (_Float16)v.y; l[1] = (_Float16)(v.y - (float)h[1]);
    h[2] = (_Float16)v.z; l[2] = (_Float16)(v.z - (float)h[2]);
    h[3] = (_Float16)v.w; l[3] = (_Float16)(v.w - (float)h[3]);
    *(v4h*)(hi + row * LDSTRIDE + col) = h;
    *(v4h*)(lo + row * LDSTRIDE + col) = l;
  }
}

// ---------------- kernel 1: squared norms ----------------
__global__ __launch_bounds__(256) void norms_kernel(const float* __restrict__ P,
                                                    float* __restrict__ norms, int s0) {
  int g = blockIdx.x * 256 + threadIdx.x;     // chunk-local point id
  int c = g >> 10, p = g & 1023;
  const float4* r = (const float4*)(P + ((size_t)(s0 + c) << 17) + ((size_t)p << 7));
  float s = 0.f;
#pragma unroll
  for (int i = 0; i < 32; ++i) {
    float4 v = r[i];
    s += v.x * v.x + v.y * v.y + v.z * v.z + v.w * v.w;
  }
  norms[(c << 10) + p] = s;
}

// ---------------- kernel 2: D^2 via split-f16 WMMA Gram ----------------
__global__ __launch_bounds__(256) void gram_kernel(const float* __restrict__ P,
                                                   const float* __restrict__ norms,
                                                   float* __restrict__ D, int s0) {
  __shared__ __align__(16) unsigned char lds[LDS_BYTES];
  _Float16* Ahi = (_Float16*)(lds + AHI_OFF);
  _Float16* Alo = (_Float16*)(lds + ALO_OFF);
  _Float16* Bhi = (_Float16*)(lds + BHI_OFF);
  _Float16* Blo = (_Float16*)(lds + BLO_OFF);
  float*    nA  = (float*)(lds + NA_OFF);
  float*    nB  = (float*)(lds + NB_OFF);

  const int tid   = threadIdx.x;
  const int c     = blockIdx.y;               // chunk-local sample
  const int Mbase = blockIdx.x << 7;          // 128-row block
  const float* Ps = P + ((size_t)(s0 + c) << 17);

#if USE_TDM
  // --- Tensor Data Mover: DMA the fp32 A tile (128x128, contiguous rows) to LDS stage ---
  if (tid < 32) {                             // one wave issues (TDM ignores EXEC; per-wave op)
    uint64_t ga = (uint64_t)(uintptr_t)(const void*)(Ps + ((size_t)Mbase << 7));
    uint32_t la = (uint32_t)(uintptr_t)(void*)(lds + STG_OFF);   // low 32 bits = LDS byte offset
    v4u g0; v8i g1; v4i z4 = {0, 0, 0, 0};
    g0[0] = 1u;                                   // count=1, user descriptor
    g0[1] = la;                                   // lds_addr
    g0[2] = (uint32_t)ga;                         // global_addr[31:0]
    g0[3] = ((uint32_t)(ga >> 32) & 0x01FFFFFFu) | (2u << 30);  // addr[56:32] | type=2
    g1[0] = (int)(2u << 16);                      // data_size=4B, no pad, no multicast
    g1[1] = (int)(128u << 16);                    // tensor_dim0=128 (bits 79:48 low half)
    g1[2] = (int)(128u << 16);                    // tensor_dim1=128
    g1[3] = (int)(128u << 16);                    // tile_dim0=128
    g1[4] = 128;                                  // tile_dim1=128, tile_dim2=0
    g1[5] = 128;                                  // tensor_dim0_stride=128
    g1[6] = 0; g1[7] = 0;                         // tensor_dim1_stride=0 (2D)
#if defined(__clang_major__) && __clang_major__ >= 23
    v8i z8 = {0, 0, 0, 0, 0, 0, 0, 0};
    __builtin_amdgcn_tensor_load_to_lds(g0, g1, z4, z4, z8, 0);
#else
    __builtin_amdgcn_tensor_load_to_lds(g0, g1, z4, z4, 0);
#endif
    __builtin_amdgcn_s_wait_tensorcnt(0);
  }
  __syncthreads();                                // stage ready for everyone
  planes_from_f32((const float*)(lds + STG_OFF), 128, Ahi, Alo, tid);
#else
  planes_from_f32(Ps + ((size_t)Mbase << 7), 128, Ahi, Alo, tid);
#endif
  if (tid < 128) nA[tid] = norms[(c << 10) + Mbase + tid];
  __syncthreads();                                // A planes done; stage region free for B

  const int lane = tid & 31, w = tid >> 5;
  const int wm = w >> 1, wn = w & 1;              // 4x2 wave grid -> 32x64 per wave
  const int r15 = lane & 15, lh = lane >> 4;

  for (int nb = 0; nb < 8; ++nb) {
    const int Nbase = nb << 7;
    planes_from_f32(Ps + ((size_t)Nbase << 7), 128, Bhi, Blo, tid);
    if (tid < 128) nB[tid] = norms[(c << 10) + Nbase + tid];
    __syncthreads();

    v8f acc[2][4];
#pragma unroll
    for (int mi = 0; mi < 2; ++mi)
#pragma unroll
      for (int ni = 0; ni < 4; ++ni) acc[mi][ni] = (v8f)(0.0f);

#pragma unroll
    for (int k0 = 0; k0 < 128; k0 += 32) {
      v16h ah[2], al[2];
#pragma unroll
      for (int mi = 0; mi < 2; ++mi) {
        const _Float16* pr = Ahi + (size_t)((wm << 5) + (mi << 4) + r15) * LDSTRIDE;
        const _Float16* pl = Alo + (size_t)((wm << 5) + (mi << 4) + r15) * LDSTRIDE;
        // A layout: lane holds 16 consecutive K in [lh*16, +16), matching B
        ah[mi] = cat8(*(const v8h*)(pr + k0 + (lh << 4)),
                      *(const v8h*)(pr + k0 + 8 + (lh << 4)));
        al[mi] = cat8(*(const v8h*)(pl + k0 + (lh << 4)),
                      *(const v8h*)(pl + k0 + 8 + (lh << 4)));
      }
#pragma unroll
      for (int ni = 0; ni < 4; ++ni) {
        const _Float16* pbh = Bhi + (size_t)((wn << 6) + (ni << 4) + r15) * LDSTRIDE + k0 + (lh << 4);
        const _Float16* pbl = Blo + (size_t)((wn << 6) + (ni << 4) + r15) * LDSTRIDE + k0 + (lh << 4);
        // B layout: lane holds 16 consecutive K for column N=lane&15, halves by lane>>4
        v16h bh = cat8(*(const v8h*)(pbh), *(const v8h*)(pbh + 8));
        v16h bl = cat8(*(const v8h*)(pbl), *(const v8h*)(pbl + 8));
#pragma unroll
        for (int mi = 0; mi < 2; ++mi) {
          acc[mi][ni] = wmma_f16(ah[mi], bh, acc[mi][ni]);   // hi*hi
          acc[mi][ni] = wmma_f16(ah[mi], bl, acc[mi][ni]);   // hi*lo
          acc[mi][ni] = wmma_f16(al[mi], bh, acc[mi][ni]);   // lo*hi
        }
      }
    }

    // epilogue: D^2 = n_m + n_n - 2*G   (C/D layout: VGPR r -> M = r + 8*(lane>>4))
#pragma unroll
    for (int mi = 0; mi < 2; ++mi)
#pragma unroll
      for (int ni = 0; ni < 4; ++ni) {
        const int mbase_l = (wm << 5) + (mi << 4) + (lh << 3);
        const int col_l   = (wn << 6) + (ni << 4) + r15;
        const float nc = nB[col_l];
#pragma unroll
        for (int r = 0; r < 8; ++r) {
          const int m = mbase_l + r;
          float val = fmaxf(nA[m] + nc - 2.0f * acc[mi][ni][r], 0.0f);
          D[((size_t)c << 20) + ((size_t)(Mbase + m) << 10) + (size_t)(Nbase + col_l)] = val;
        }
      }
    __syncthreads();                              // B planes reusable next iter
  }
}

// ---------------- kernel 3: Prim's MST over precomputed D^2 ----------------
__device__ inline unsigned long long shflx64(unsigned long long v, int m) {
  unsigned lo = (unsigned)v, hi = (unsigned)(v >> 32);
  lo = __shfl_xor(lo, m, 32);
  hi = __shfl_xor(hi, m, 32);
  return ((unsigned long long)hi << 32) | lo;
}

__global__ __launch_bounds__(256) void prim_kernel(const float* __restrict__ D,
                                                   float* __restrict__ means, int s0) {
  __shared__ unsigned long long red[8];
  __shared__ unsigned long long bestS;
  const int tid  = threadIdx.x;
  const int c    = blockIdx.x;
  const float* Ds = D + ((size_t)c << 20);
  const int base = tid << 2;                      // each thread owns 4 points

  float4 md = *(const float4*)(Ds + base);        // distances^2 to point 0
  int visited = (tid == 0) ? 1 : 0;               // point 0 visited
  float dsum = 0.0f;

  for (int step = 0; step < N_PTS - 1; ++step) {
    // local masked argmin, packed (distbits<<32)|idx: min => smallest dist, then idx
    unsigned long long best = ~0ull;
    float e0 = md.x, e1 = md.y, e2 = md.z, e3 = md.w;
    if (!(visited & 1)) { unsigned long long cd = ((unsigned long long)__float_as_uint(e0) << 32) | (unsigned)(base + 0); best = best < cd ? best : cd; }
    if (!(visited & 2)) { unsigned long long cd = ((unsigned long long)__float_as_uint(e1) << 32) | (unsigned)(base + 1); best = best < cd ? best : cd; }
    if (!(visited & 4)) { unsigned long long cd = ((unsigned long long)__float_as_uint(e2) << 32) | (unsigned)(base + 2); best = best < cd ? best : cd; }
    if (!(visited & 8)) { unsigned long long cd = ((unsigned long long)__float_as_uint(e3) << 32) | (unsigned)(base + 3); best = best < cd ? best : cd; }
#pragma unroll
    for (int off = 16; off; off >>= 1) {
      unsigned long long o = shflx64(best, off);
      best = best < o ? best : o;
    }
    if ((tid & 31) == 0) red[tid >> 5] = best;
    __syncthreads();
    if (tid == 0) {
      unsigned long long b = red[0];
#pragma unroll
      for (int i = 1; i < 8; ++i) b = b < red[i] ? b : red[i];
      bestS = b;
    }
    __syncthreads();
    const unsigned long long b = bestS;
    const int j = (int)(b & 0xFFFFFFFFu);
    if (tid == 0) dsum += sqrtf(__uint_as_float((unsigned)(b >> 32)));
    if ((j >> 2) == tid) visited |= 1 << (j & 3);
    const float4 r = *(const float4*)(Ds + ((size_t)j << 10) + base);
    md.x = fminf(md.x, r.x); md.y = fminf(md.y, r.y);
    md.z = fminf(md.z, r.z); md.w = fminf(md.w, r.w);
    // red/bestS reuse is safe: rewrites happen only after the next iteration's barriers
  }
  if (tid == 0) means[s0 + c] = dsum * (1.0f / (float)(N_PTS - 1));
}

// ---------------- kernel 4: deterministic final reduction ----------------
__global__ __launch_bounds__(128) void final_kernel(const float* __restrict__ means,
                                                    float* __restrict__ out) {
  __shared__ float buf[128];
  const int tid = threadIdx.x;
  buf[tid] = means[tid];
  __syncthreads();
#pragma unroll
  for (int s = 64; s; s >>= 1) {
    if (tid < s) buf[tid] += buf[tid + s];
    __syncthreads();
  }
  if (tid == 0) out[0] = buf[0] * (1.0f / (float)(NBATCH * 2));   // / (B * n_target_dims)
}

__global__ void zero_kernel(float* out) { out[0] = 0.0f; }

// ---------------- host ----------------
extern "C" void kernel_launch(void* const* d_in, const int* in_sizes, int n_in,
                              void* d_out, int out_size, void* d_ws, size_t ws_size,
                              hipStream_t stream) {
  (void)in_sizes; (void)n_in; (void)out_size;
  const float* P = (const float*)d_in[0];
  float* out = (float*)d_out;
  float* ws  = (float*)d_ws;

  // ws layout (floats): [0,128) per-sample means | [128, 128+CH*1024) norms | D^2: CH*1M
  const size_t wsf = ws_size / 4;
  long avail = (long)wsf - 128;
  int CH = (avail > 0) ? (int)(avail / (1048576 + 1024)) : 0;
  if (CH > NBATCH) CH = NBATCH;
  if (CH < 1) { zero_kernel<<<1, 1, 0, stream>>>(out); return; }

  float* means = ws;
  float* norms = ws + 128;
  float* D     = ws + 128 + (size_t)CH * 1024;

  for (int s0 = 0; s0 < NBATCH; s0 += CH) {
    const int n = (NBATCH - s0 < CH) ? (NBATCH - s0) : CH;
    norms_kernel<<<dim3(n * 4), 256, 0, stream>>>(P, norms, s0);
    gram_kernel <<<dim3(8, n),  256, 0, stream>>>(P, norms, D, s0);
    prim_kernel <<<dim3(n),     256, 0, stream>>>(D, means, s0);
  }
  final_kernel<<<1, 128, 0, stream>>>(means, out);
}
